// GAT_48473000903497
// MI455X (gfx1250) — compile-verified, matrix operand order checked by reading, not run on hardware
//
#include <hip/hip_runtime.h>
#include <hip/hip_bf16.h>
#include <math.h>

// Problem constants (match reference)
#define NNODES   20000
#define INDIM    256
#define NHEAD    3
#define DHEAD    64
#define HD       (NHEAD * DHEAD)   // 192
#define NEDGE    640000
#define NEG_SLOPE 0.2f

#define MTILES   (NNODES / 16)     // 1250
#define NTILES   (HD / 16)         // 12
#define KSTEPS   (INDIM / 32)      // 8

typedef __attribute__((ext_vector_type(16))) __bf16 v16bf;
typedef __attribute__((ext_vector_type(8)))  __bf16 v8bf;
typedef __attribute__((ext_vector_type(8)))  float  v8f;

// float -> bf16 bits, round-to-nearest-even (integer ops only)
__device__ inline unsigned short f2bf_bits(float f) {
  unsigned u = __float_as_uint(f);
  return (unsigned short)((u + 0x7FFFu + ((u >> 16) & 1u)) >> 16);
}

// signed/unsigned-int trick for float atomic max (init must be -inf)
__device__ inline void atomicMaxF(float* addr, float val) {
  if (val >= 0.0f) atomicMax((int*)addr, __float_as_int(val));
  else             atomicMin((unsigned int*)addr, __float_as_uint(val));
}

// K index inside a 16x32 bf16 A fragment for element j of lane-group `half`
__device__ inline int frag_kidx(int j, int half) {
  return (j < 8) ? (half * 8 + j) : (16 + half * 8 + (j - 8));
}

// ---------------------------------------------------------------------------
// Kernel 0: init accumulators. outAcc=0, m=-inf, denom=0, d_out <- fc_b
// ---------------------------------------------------------------------------
__global__ void gat_init(float* __restrict__ m, float* __restrict__ denom,
                         float* __restrict__ outAcc,
                         const float* __restrict__ fc_b,
                         float* __restrict__ dout) {
  int i = blockIdx.x * blockDim.x + threadIdx.x;
  if (i < NNODES * HD)    outAcc[i] = 0.0f;
  if (i < NNODES * NHEAD) { m[i] = -INFINITY; denom[i] = 0.0f; }
  if (i == 0) { dout[0] = fc_b[0]; dout[1] = fc_b[1]; }
}

// ---------------------------------------------------------------------------
// Kernel 1a: feat (f32) -> featbf (bf16), one-pass streaming conversion
// ---------------------------------------------------------------------------
__global__ void cvt_feat_bf16(const float* __restrict__ feat,
                              unsigned short* __restrict__ featbf) {
  int i = blockIdx.x * blockDim.x + threadIdx.x;
  if (i < NNODES * INDIM) featbf[i] = f2bf_bits(feat[i]);
}

// ---------------------------------------------------------------------------
// Kernel 1b: W (f32, [K,HD] row-major) -> B fragment image in the exact
// per-lane v16bf layout WMMA consumes:
//   Bimg[(((tn*KSTEPS + ks)*32 + lane)*16 + j] =
//       bf16( W[(ks*32 + kidx(j,half))*HD + tn*16 + (lane&15)] )
// 96 KB total -> L2 resident; GEMM reads it with contiguous b128 loads.
// ---------------------------------------------------------------------------
__global__ void build_b_image(const float* __restrict__ W,
                              unsigned short* __restrict__ Bimg) {
  int i = blockIdx.x * blockDim.x + threadIdx.x;   // ((tn*8+ks)*32+lane)*16+j
  if (i >= NTILES * KSTEPS * 32 * 16) return;
  int j    = i & 15;
  int lane = (i >> 4) & 31;
  int ks   = (i >> 9) % KSTEPS;
  int tn   = (i >> 9) / KSTEPS;
  int half = lane >> 4;
  int l16  = lane & 15;
  int k    = ks * 32 + frag_kidx(j, half);
  Bimg[i]  = f2bf_bits(W[(size_t)k * HD + tn * 16 + l16]);
}

// ---------------------------------------------------------------------------
// Kernel 2: h = feat @ W via v_wmma_f32_16x16x32_bf16.
// One wave per 16-row strip of h; A fragments (full K=256) held in registers
// and reused across all 12 N-tiles -> 96 WMMAs per wave, inner k fully
// unrolled (8 static v_wmma). All operand loads are b128, zero conversions.
// ---------------------------------------------------------------------------
__global__ void gat_gemm_wmma(const unsigned short* __restrict__ featbf,
                              const unsigned short* __restrict__ Bimg,
                              float* __restrict__ h) {
  const int lane = threadIdx.x & 31;
  const int wave = threadIdx.x >> 5;
  const int tm   = blockIdx.x * 8 + wave;          // m-tile (wave-uniform)
  if (tm >= MTILES) return;                        // wave-uniform exit
  const int half = lane >> 4;
  const int l16  = lane & 15;
  const int arow = tm * 16 + l16;

  // Load all A fragments for K=256: 8 fragments x (2 x 16B loads)
  v16bf a[KSTEPS];
#pragma unroll
  for (int ks = 0; ks < KSTEPS; ++ks) {
    const unsigned short* ap = featbf + (size_t)arow * INDIM + ks * 32 + half * 8;
    v8bf lo = *(const v8bf*)(ap);        // K = half*8 .. +7
    v8bf hi = *(const v8bf*)(ap + 16);   // K = 16+half*8 .. +7
    a[ks] = __builtin_shufflevector(lo, hi, 0, 1, 2, 3, 4, 5, 6, 7,
                                            8, 9, 10, 11, 12, 13, 14, 15);
  }

  for (int tn = 0; tn < NTILES; ++tn) {
    v8f c = {};
    const unsigned short* bbase =
        Bimg + ((size_t)tn * KSTEPS * 32 + lane) * 16;
#pragma unroll
    for (int ks = 0; ks < KSTEPS; ++ks) {
      v16bf b = *(const v16bf*)(bbase + (size_t)ks * 32 * 16);  // 32B/lane
      c = __builtin_amdgcn_wmma_f32_16x16x32_bf16(false, a[ks], false, b,
                                                  (short)0, c, false, false);
    }
    // C/D layout: VGPR r -> row r (lanes 0-15) / row r+8 (lanes 16-31)
    float* hout = h + (size_t)(tm * 16 + half * 8) * HD + tn * 16 + l16;
#pragma unroll
    for (int r = 0; r < 8; ++r) hout[(size_t)r * HD] = c[r];
  }
}

// ---------------------------------------------------------------------------
// Kernel 3: el[n,h] = <h[n,h,:], attn_l[h,:]>, er likewise. 1 thread / (n,h)
// ---------------------------------------------------------------------------
__global__ void gat_el_er(const float* __restrict__ h,
                          const float* __restrict__ attn_l,
                          const float* __restrict__ attn_r,
                          float* __restrict__ el, float* __restrict__ er) {
  int i = blockIdx.x * blockDim.x + threadIdx.x;   // n*NHEAD + hh
  if (i >= NNODES * NHEAD) return;
  int n = i / NHEAD, hh = i % NHEAD;
  const float* hr = h + (size_t)n * HD + hh * DHEAD;
  const float* al = attn_l + hh * DHEAD;
  const float* ar = attn_r + hh * DHEAD;
  float sl = 0.0f, sr = 0.0f;
#pragma unroll 8
  for (int d = 0; d < DHEAD; ++d) {
    float v = hr[d];
    sl += v * al[d];
    sr += v * ar[d];
  }
  el[i] = sl;
  er[i] = sr;
}

__device__ inline float edge_logit(const float* el, const float* er,
                                   int s, int d, int hh) {
  float x = el[s * NHEAD + hh] + er[d * NHEAD + hh];
  return x > 0.0f ? x : NEG_SLOPE * x;
}

// ---------------------------------------------------------------------------
// Kernel 4: segment max over destination (edge softmax, stable)
// ---------------------------------------------------------------------------
__global__ void gat_edge_max(const int* __restrict__ src,
                             const int* __restrict__ dst,
                             const float* __restrict__ el,
                             const float* __restrict__ er,
                             float* __restrict__ m) {
  int i = blockIdx.x * blockDim.x + threadIdx.x;   // e*NHEAD + hh
  if (i >= NEDGE * NHEAD) return;
  int e = i / NHEAD, hh = i % NHEAD;
  int s = src[e], d = dst[e];
  atomicMaxF(&m[d * NHEAD + hh], edge_logit(el, er, s, d, hh));
}

// ---------------------------------------------------------------------------
// Kernel 5: denom[v,h] = sum exp(e - m)
// ---------------------------------------------------------------------------
__global__ void gat_edge_denom(const int* __restrict__ src,
                               const int* __restrict__ dst,
                               const float* __restrict__ el,
                               const float* __restrict__ er,
                               const float* __restrict__ m,
                               float* __restrict__ denom) {
  int i = blockIdx.x * blockDim.x + threadIdx.x;
  if (i >= NEDGE * NHEAD) return;
  int e = i / NHEAD, hh = i % NHEAD;
  int s = src[e], d = dst[e];
  float x  = edge_logit(el, er, s, d, hh);
  float ex = __expf(x - m[d * NHEAD + hh]);
  atomicAdd(&denom[d * NHEAD + hh], ex);
}

// ---------------------------------------------------------------------------
// Kernel 6: out[v,h,:] += alpha * h[u,h,:]  (scatter-add; h is L2-resident)
// ---------------------------------------------------------------------------
__global__ void gat_edge_aggregate(const int* __restrict__ src,
                                   const int* __restrict__ dst,
                                   const float* __restrict__ el,
                                   const float* __restrict__ er,
                                   const float* __restrict__ m,
                                   const float* __restrict__ denom,
                                   const float* __restrict__ h,
                                   float* __restrict__ outAcc) {
  int i = blockIdx.x * blockDim.x + threadIdx.x;   // e*NHEAD + hh
  if (i >= NEDGE * NHEAD) return;
  int e = i / NHEAD, hh = i % NHEAD;
  int s = src[e], d = dst[e];
  float x     = edge_logit(el, er, s, d, hh);
  float alpha = __expf(x - m[d * NHEAD + hh]) / denom[d * NHEAD + hh];
  const float* hs = h      + (size_t)s * HD + hh * DHEAD;
  float*       ov = outAcc + (size_t)d * HD + hh * DHEAD;
  __builtin_prefetch(hs, 0, 0);                    // global_prefetch_b8
#pragma unroll 8
  for (int dd = 0; dd < DHEAD; ++dd) {
    atomicAdd(&ov[dd], alpha * hs[dd]);
  }
}

// ---------------------------------------------------------------------------
// Kernel 7: relu(out + bias) dotted with fc_w[:,0] / fc_w[:,1] -> d_out[0..1]
// wave32 shuffle reduction, then 2 atomics per wave.
// ---------------------------------------------------------------------------
__global__ void gat_final(const float* __restrict__ outAcc,
                          const float* __restrict__ bias,
                          const float* __restrict__ fc_w,
                          float* __restrict__ dout) {
  int i = blockIdx.x * blockDim.x + threadIdx.x;
  float p0 = 0.0f, p1 = 0.0f;
  if (i < NNODES * HD) {
    float v = outAcc[i] + bias[i % HD];
    v = v > 0.0f ? v : 0.0f;
    p0 = v * fc_w[(size_t)i * 2 + 0];
    p1 = v * fc_w[(size_t)i * 2 + 1];
  }
#pragma unroll
  for (int off = 16; off > 0; off >>= 1) {
    p0 += __shfl_xor(p0, off, 32);
    p1 += __shfl_xor(p1, off, 32);
  }
  if ((threadIdx.x & 31) == 0) {
    atomicAdd(&dout[0], p0);
    atomicAdd(&dout[1], p1);
  }
}

// ---------------------------------------------------------------------------
extern "C" void kernel_launch(void* const* d_in, const int* in_sizes, int n_in,
                              void* d_out, int out_size, void* d_ws, size_t ws_size,
                              hipStream_t stream) {
  const float* feat   = (const float*)d_in[0];
  const float* W      = (const float*)d_in[1];
  const float* attn_l = (const float*)d_in[2];
  const float* attn_r = (const float*)d_in[3];
  const float* gbias  = (const float*)d_in[4];
  const float* fc_w   = (const float*)d_in[5];
  const float* fc_b   = (const float*)d_in[6];
  const int*   src    = (const int*)d_in[7];
  const int*   dst    = (const int*)d_in[8];
  float* dout = (float*)d_out;

  // workspace layout: ~42 MB total
  float* ws     = (float*)d_ws;
  float* h      = ws;                              // N*HD    = 3,840,000 f32
  float* el     = h      + (size_t)NNODES * HD;    // N*NHEAD =    60,000 f32
  float* er     = el     + (size_t)NNODES * NHEAD;
  float* m      = er     + (size_t)NNODES * NHEAD;
  float* denom  = m      + (size_t)NNODES * NHEAD;
  float* outAcc = denom  + (size_t)NNODES * NHEAD; // N*HD f32
  unsigned short* featbf = (unsigned short*)(outAcc + (size_t)NNODES * HD); // N*INDIM bf16
  unsigned short* Bimg   = featbf + (size_t)NNODES * INDIM;                 // 49,152 bf16

  const int B = 256;
  const int nhdBlocks  = (NNODES * HD + B - 1) / B;            // 15000
  const int cvtBlocks  = (NNODES * INDIM + B - 1) / B;         // 20000
  const int bimgBlocks = (NTILES * KSTEPS * 32 * 16 + B - 1) / B; // 192
  const int gemmBlocks = (MTILES + 7) / 8;                     // 157
  const int nhBlocks   = (NNODES * NHEAD + B - 1) / B;         // 235
  const int ehBlocks   = (NEDGE * NHEAD + B - 1) / B;          // 7500

  gat_init<<<nhdBlocks, B, 0, stream>>>(m, denom, outAcc, fc_b, dout);
  cvt_feat_bf16<<<cvtBlocks, B, 0, stream>>>(feat, featbf);
  build_b_image<<<bimgBlocks, B, 0, stream>>>(W, Bimg);
  gat_gemm_wmma<<<gemmBlocks, B, 0, stream>>>(featbf, Bimg, h);
  gat_el_er<<<nhBlocks, B, 0, stream>>>(h, attn_l, attn_r, el, er);
  gat_edge_max<<<ehBlocks, B, 0, stream>>>(src, dst, el, er, m);
  gat_edge_denom<<<ehBlocks, B, 0, stream>>>(src, dst, el, er, m, denom);
  gat_edge_aggregate<<<ehBlocks, B, 0, stream>>>(src, dst, el, er, m, denom, h, outAcc);
  gat_final<<<nhdBlocks, B, 0, stream>>>(outAcc, gbias, fc_w, dout);
}
